// ICLTAdapter_3367254360725
// MI455X (gfx1250) — compile-verified, matrix operand order checked by reading, not run on hardware
//
#include <hip/hip_runtime.h>

typedef __attribute__((ext_vector_type(16))) __bf16 v16bf;
typedef __attribute__((ext_vector_type(8)))  __bf16 v8bf;
typedef __attribute__((ext_vector_type(4)))  __bf16 v4bf;
typedef __attribute__((ext_vector_type(8)))  float  v8f;

// ---------- bf16 <-> f32 helpers (bit-exact RNE, storage-type only) ----------
__device__ __forceinline__ __bf16 f2bf(float f) {
  union { float f; unsigned u; } in; in.f = f;
  unsigned r = (in.u + 0x7FFFu + ((in.u >> 16) & 1u)) >> 16;
  union { unsigned short s; __bf16 b; } out; out.s = (unsigned short)r;
  return out.b;
}
__device__ __forceinline__ float bf2f(__bf16 b) {
  union { __bf16 b; unsigned short s; } in; in.b = b;
  union { unsigned u; float f; } out; out.u = ((unsigned)in.s) << 16;
  return out.f;
}

// ---------- conversion kernels ----------
__global__ __launch_bounds__(256)
void k_cvt_flat4(const float* __restrict__ in, __bf16* __restrict__ out, size_t n4) {
  size_t i = (size_t)blockIdx.x * 256 + threadIdx.x;
  if (i >= n4) return;
  float4 f = ((const float4*)in)[i];
  v4bf o;
  o[0] = f2bf(f.x); o[1] = f2bf(f.y); o[2] = f2bf(f.z); o[3] = f2bf(f.w);
  ((v4bf*)out)[i] = o;
}

// out[c*rows + r] = (bf16) in[r*cols + c]
__global__ __launch_bounds__(256)
void k_transpose_cvt(const float* __restrict__ in, __bf16* __restrict__ out,
                     int rows, int cols) {
  __shared__ float tile[32][33];
  const int tx = threadIdx.x, ty = threadIdx.y;
  const int c0 = blockIdx.x * 32, r0 = blockIdx.y * 32;
#pragma unroll
  for (int j = 0; j < 32; j += 8)
    tile[ty + j][tx] = in[(size_t)(r0 + ty + j) * cols + (c0 + tx)];
  __syncthreads();
#pragma unroll
  for (int j = 0; j < 32; j += 8)
    out[(size_t)(c0 + ty + j) * rows + (r0 + tx)] = f2bf(tile[tx][ty + j]);
}

// Ub[d*(K*R) + k*R + r] = U[k*D*R + d*R + r]   (coalesced both sides)
__global__ __launch_bounds__(256)
void k_cvt_U(const float* __restrict__ U, __bf16* __restrict__ Ub,
             int D, int R, int K) {
  int i = blockIdx.x * 256 + threadIdx.x;       // over K*D*R
  int r = i % R;
  int d = (i / R) % D;
  int k = i / (R * D);
  Ub[(size_t)d * (K * R) + k * R + r] = f2bf(U[i]);
}

// ---------- WMMA GEMM: C = epilogue(A[M,Kd] * Bt[N,Kd]^T) ----------
constexpr int BM = 128, BN = 128, BK = 64;
constexpr int LDT = 72;               // LDS row stride in halfs (144B, 16B aligned,
                                      // 36*row mod 64 banks -> conflict-free frags)
constexpr int TILEE = BM * LDT;       // halfs per buffer (A and B identical)

// Async global->LDS copy of a 64B per-thread chunk (4 x b128).
// GVS addressing: uniform SGPR base + per-lane 32-bit byte offset.
// The offset: immediate advances both the LDS and global address (ISA 8.4.4).
#define ASYNC_CHUNK64(ldsaddr, vbyteoff, sbase)                              \
  asm volatile("global_load_async_to_lds_b128 %0, %1, %2 offset:0\n\t"       \
               "global_load_async_to_lds_b128 %0, %1, %2 offset:16\n\t"      \
               "global_load_async_to_lds_b128 %0, %1, %2 offset:32\n\t"      \
               "global_load_async_to_lds_b128 %0, %1, %2 offset:48"          \
               :: "v"(ldsaddr), "v"(vbyteoff), "s"(sbase) : "memory")

// EPI 0: relu(c + bias[col]) -> bf16
// EPI 1: c * alpha[row*8 + (col>>6)] -> bf16
// EPI 2: c -> f32
template <int EPI>
__global__ __launch_bounds__(256)
void k_wmma_gemm(const __bf16* __restrict__ A, const __bf16* __restrict__ Bt,
                 const float* __restrict__ bias, const float* __restrict__ alpha,
                 void* __restrict__ Cout, int M, int N, int Kd) {
  __shared__ __align__(16) __bf16 As[2 * TILEE];
  __shared__ __align__(16) __bf16 Bs[2 * TILEE];

  const int tid     = threadIdx.x;
  const int lane    = tid & 31;
  const int wave    = tid >> 5;
  const int wm      = (wave >> 1) * 32;   // 4 waves along M
  const int wn      = (wave & 1) * 64;    // 2 waves along N
  const int blockM  = blockIdx.y * BM;
  const int blockN  = blockIdx.x * BN;
  const int laneRow = lane & 15;
  const int seg     = lane >> 4;          // K-half select per ISA fragment layout

  const int ldRow = tid >> 1;             // 128 rows, 2 threads/row
  const int ldCol = (tid & 1) * 32;       // 32 halfs (64B) per thread

  v8f zero = {};
  v8f acc[2][4];
#pragma unroll
  for (int i = 0; i < 2; ++i)
#pragma unroll
    for (int j = 0; j < 4; ++j) acc[i][j] = zero;

  // per-thread LDS byte addresses (low 32 bits of generic pointer = LDS offset)
  const unsigned ldsA[2] = {
      (unsigned)(uintptr_t)&As[ldRow * LDT + ldCol],
      (unsigned)(uintptr_t)&As[TILEE + ldRow * LDT + ldCol]};
  const unsigned ldsB[2] = {
      (unsigned)(uintptr_t)&Bs[ldRow * LDT + ldCol],
      (unsigned)(uintptr_t)&Bs[TILEE + ldRow * LDT + ldCol]};

  // per-thread global byte offsets + uniform SGPR bases
  const unsigned aByte = (unsigned)((ldRow * Kd + ldCol) * 2);
  const unsigned bByte = aByte;  // same pattern for Bt
  const unsigned long long aS = (unsigned long long)(uintptr_t)(A  + (size_t)blockM * Kd);
  const unsigned long long bS = (unsigned long long)(uintptr_t)(Bt + (size_t)blockN * Kd);

  const int nIter = Kd / BK;

  // prologue: stream tile 0 into buffer 0 (8 async instructions / wave)
  ASYNC_CHUNK64(ldsA[0], aByte, aS);
  ASYNC_CHUNK64(ldsB[0], bByte, bS);

  int cur = 0;
  for (int kk = 0; kk < nIter; ++kk) {
    if (kk + 1 < nIter) {
      // stream next tile into alternate buffer; it fills while we compute
      const unsigned ao = aByte + (unsigned)((kk + 1) * BK * 2);
      const unsigned bo = bByte + (unsigned)((kk + 1) * BK * 2);
      ASYNC_CHUNK64(ldsA[cur ^ 1], ao, aS);
      ASYNC_CHUNK64(ldsB[cur ^ 1], bo, bS);
      // 8 newest outstanding = next tile; <=8 left -> current tile resident
      asm volatile("s_wait_asynccnt 0x8" ::: "memory");
    } else {
      asm volatile("s_wait_asynccnt 0x0" ::: "memory");
    }
    __syncthreads();   // all waves' async waits done -> tile kk visible to all

    const __bf16* aT = As + cur * TILEE;
    const __bf16* bT = Bs + cur * TILEE;
#pragma unroll
    for (int ks = 0; ks < 2; ++ks) {       // two 16x16x32 K-steps per staged tile
      const int kcol = ks * 32 + seg * 8;
      v16bf afrag[2], bfrag[4];
#pragma unroll
      for (int ti = 0; ti < 2; ++ti) {
        const __bf16* p = aT + (wm + ti * 16 + laneRow) * LDT + kcol;
        v8bf lo = *(const v8bf*)p;
        v8bf hi = *(const v8bf*)(p + 16);
        afrag[ti] = __builtin_shufflevector(lo, hi, 0,1,2,3,4,5,6,7,8,9,10,11,12,13,14,15);
      }
#pragma unroll
      for (int tj = 0; tj < 4; ++tj) {
        const __bf16* p = bT + (wn + tj * 16 + laneRow) * LDT + kcol;
        v8bf lo = *(const v8bf*)p;
        v8bf hi = *(const v8bf*)(p + 16);
        bfrag[tj] = __builtin_shufflevector(lo, hi, 0,1,2,3,4,5,6,7,8,9,10,11,12,13,14,15);
      }
#pragma unroll
      for (int ti = 0; ti < 2; ++ti)
#pragma unroll
        for (int tj = 0; tj < 4; ++tj)
          acc[ti][tj] = __builtin_amdgcn_wmma_f32_16x16x32_bf16(
              false, afrag[ti], false, bfrag[tj], (short)0, acc[ti][tj], false, false);
    }
    __syncthreads();   // all reads done before next iteration overwrites buf[cur]
    cur ^= 1;
  }

#pragma unroll
  for (int ti = 0; ti < 2; ++ti) {
    const int rowBase = blockM + wm + ti * 16 + seg * 8;
#pragma unroll
    for (int tj = 0; tj < 4; ++tj) {
      const int col = blockN + wn + tj * 16 + laneRow;
      float bv = 0.0f;
      if constexpr (EPI == 0) bv = bias[col];
#pragma unroll
      for (int v = 0; v < 8; ++v) {
        const int row = rowBase + v;
        float c = acc[ti][tj][v];
        if constexpr (EPI == 0) {
          float h = c + bv;
          h = h > 0.0f ? h : 0.0f;
          ((__bf16*)Cout)[(size_t)row * N + col] = f2bf(h);
        } else if constexpr (EPI == 1) {
          float a = alpha[(size_t)row * 8 + (col >> 6)];
          ((__bf16*)Cout)[(size_t)row * N + col] = f2bf(c * a);
        } else {
          ((float*)Cout)[(size_t)row * N + col] = c;
        }
      }
    }
  }
}

// ---------- router tail: logits = h*W2 + b2 ; alpha = softmax_K ----------
__global__ __launch_bounds__(256)
void k_router(const __bf16* __restrict__ h, const float* __restrict__ W2,
              const float* __restrict__ b2, float* __restrict__ alpha, int D) {
  const int token = blockIdx.x * 8 + (threadIdx.x >> 5); // one wave32 per token
  const int lane  = threadIdx.x & 31;
  const __bf16* hr = h + (size_t)token * D;

  float acc[8];
#pragma unroll
  for (int k = 0; k < 8; ++k) acc[k] = 0.0f;

  for (int e = lane; e < D; e += 32) {
    float hv = bf2f(hr[e]);
    const float4* w2 = (const float4*)(W2 + (size_t)e * 8);
    float4 w0 = w2[0], w1 = w2[1];
    acc[0] += hv * w0.x; acc[1] += hv * w0.y; acc[2] += hv * w0.z; acc[3] += hv * w0.w;
    acc[4] += hv * w1.x; acc[5] += hv * w1.y; acc[6] += hv * w1.z; acc[7] += hv * w1.w;
  }
#pragma unroll
  for (int k = 0; k < 8; ++k) {
    acc[k] += __shfl_xor(acc[k], 16, 32);
    acc[k] += __shfl_xor(acc[k],  8, 32);
    acc[k] += __shfl_xor(acc[k],  4, 32);
    acc[k] += __shfl_xor(acc[k],  2, 32);
    acc[k] += __shfl_xor(acc[k],  1, 32);
  }
  float logit[8], m = -1e30f;
#pragma unroll
  for (int k = 0; k < 8; ++k) { logit[k] = acc[k] + b2[k]; m = fmaxf(m, logit[k]); }
  float ex[8], s = 0.0f;
#pragma unroll
  for (int k = 0; k < 8; ++k) { ex[k] = __expf(logit[k] - m); s += ex[k]; }
  float inv = 1.0f / s;
  if (lane < 8) alpha[(size_t)token * 8 + lane] = ex[lane] * inv;
}

// ---------- launcher ----------
extern "C" void kernel_launch(void* const* d_in, const int* in_sizes, int n_in,
                              void* d_out, int out_size, void* d_ws, size_t ws_size,
                              hipStream_t stream) {
  (void)in_sizes; (void)n_in; (void)out_size; (void)ws_size;
  constexpr int D = 2048, K = 8, R = 64;
  constexpr int NT = 8192;      // B*S tokens
  constexpr int KR = K * R;     // 512

  const float* x  = (const float*)d_in[0];
  const float* U  = (const float*)d_in[1];
  const float* V  = (const float*)d_in[2];
  const float* W1 = (const float*)d_in[3];
  const float* b1 = (const float*)d_in[4];
  const float* W2 = (const float*)d_in[5];
  const float* b2 = (const float*)d_in[6];
  float* out = (float*)d_out;

  char* ws = (char*)d_ws;
  size_t off = 0;
  auto take = [&](size_t bytes) {
    char* p = ws + off;
    off = (off + bytes + 255) & ~(size_t)255;
    return (void*)p;
  };
  __bf16* xb  = (__bf16*)take((size_t)NT * D * 2);   // x in bf16
  __bf16* hb  = (__bf16*)take((size_t)NT * D * 2);   // relu(xW1+b1)
  __bf16* W1t = (__bf16*)take((size_t)D * D * 2);    // W1^T
  __bf16* Vb  = (__bf16*)take((size_t)KR * D * 2);   // V flat [KR, D]
  __bf16* Ub  = (__bf16*)take((size_t)D * KR * 2);   // Ubig [D, KR]
  float*  alp = (float*)take((size_t)NT * K * 4);    // softmax weights
  __bf16* wb  = (__bf16*)take((size_t)NT * KR * 2);  // alpha ⊙ (xV^T)

  // precision conversions / layout shuffles
  k_cvt_flat4<<<(NT * D / 4) / 256, 256, 0, stream>>>(x, xb, (size_t)(NT * D / 4));
  k_cvt_flat4<<<(KR * D / 4) / 256, 256, 0, stream>>>(V, Vb, (size_t)(KR * D / 4));
  k_transpose_cvt<<<dim3(D / 32, D / 32), dim3(32, 8), 0, stream>>>(W1, W1t, D, D);
  k_cvt_U<<<(K * D * R) / 256, 256, 0, stream>>>(U, Ub, D, R, K);

  // h = relu(x @ W1 + b1)          [8192,2048] x [2048,2048]
  k_wmma_gemm<0><<<dim3(D / BN, NT / BM), 256, 0, stream>>>(xb, W1t, b1, nullptr, hb, NT, D, D);
  // alpha = softmax(h @ W2 + b2)
  k_router<<<NT / 8, 256, 0, stream>>>(hb, W2, b2, alp, D);
  // w = alpha ⊙ (x @ V^T)          [8192,2048] x [2048,512]
  k_wmma_gemm<1><<<dim3(KR / BN, NT / BM), 256, 0, stream>>>(xb, Vb, nullptr, alp, wb, NT, KR, D);
  // out = w @ Ubig^T               [8192,512] x [512,2048] -> f32
  k_wmma_gemm<2><<<dim3(D / BN, NT / BM), 256, 0, stream>>>(wb, Ub, nullptr, nullptr, out, NT, D, KR);
}
